// GL_GCNConv_9l_128h_nw_44753559224353
// MI455X (gfx1250) — compile-verified, matrix operand order checked by reading, not run on hardware
//
#include <hip/hip_runtime.h>
#include <math.h>

typedef __attribute__((ext_vector_type(16))) __bf16 v16bf;
typedef __attribute__((ext_vector_type(8)))  float  v8f;
typedef unsigned short u16;
typedef unsigned int   u32;

#define HID 128
#define BM  128
#define BN  64
#define BK  32
#define BKP 40   // padded K-stride in bf16 elems: 80 bytes -> 16B-aligned rows, bank-spread

// ---------- bf16 helpers (hi/lo split for fp32-accurate WMMA GEMM) ----------
__device__ __forceinline__ float bfh_to_f(u16 h) {
  union { u32 u; float f; } c; c.u = ((u32)h) << 16; return c.f;
}
__device__ __forceinline__ u16 f_to_bf_rne(float f) {
  union { float f; u32 u; } c; c.f = f;
  u32 r = c.u + 0x7FFFu + ((c.u >> 16) & 1u);
  return (u16)(r >> 16);
}
__device__ __forceinline__ void split_bf(float v, u16& hi, u16& lo) {
  union { float f; u32 u; } c; c.f = v;
  u16 h = (u16)(c.u >> 16);            // truncate -> residual exact in fp32
  float r = v - bfh_to_f(h);
  hi = h; lo = f_to_bf_rne(r);
}

// ---------- WMMA GEMM: C[nrows x 128] = A[nrows x 128] @ W[128 x 128] ----------
__global__ __launch_bounds__(256)
void gemm_bf16x3(const float* __restrict__ A, const float* __restrict__ W,
                 float* __restrict__ C, int nrows) {
  __shared__ u16 aHi[BM][BKP];
  __shared__ u16 aLo[BM][BKP];
  __shared__ u16 bHi[BN][BKP];   // W tile stored transposed: [n][k]
  __shared__ u16 bLo[BN][BKP];

  const int tid  = threadIdx.x;
  const int lane = tid & 31;
  const int wave = tid >> 5;           // 0..7 -> m-tile index
  const int l15  = lane & 15;
  const int kh   = (lane >> 4) << 3;   // lanes 0-15: K0..7/K16..23; lanes 16-31: K8..15/K24..31
  const int row0 = blockIdx.x * BM;
  const int col0 = blockIdx.y * BN;
  const bool fullTile = (row0 + BM) <= nrows;   // 390 of 391 M-blocks: guard-free path

  v8f acc[4];
  for (int t = 0; t < 4; ++t)
    for (int r = 0; r < 8; ++r) acc[t][r] = 0.0f;

  for (int kc = 0; kc < HID; kc += BK) {
    __syncthreads();
    // cooperative fill: A tile (rows of h), split into hi/lo bf16
    if (fullTile) {
      #pragma unroll
      for (int idx = tid; idx < BM * BK; idx += 256) {
        const int m = idx >> 5, k = idx & 31;
        float v = A[(long long)(row0 + m) * HID + (kc + k)];
        split_bf(v, aHi[m][k], aLo[m][k]);
      }
    } else {
      for (int idx = tid; idx < BM * BK; idx += 256) {
        const int m = idx >> 5, k = idx & 31;
        const int row = row0 + m;
        float v = (row < nrows) ? A[(long long)row * HID + (kc + k)] : 0.0f;
        split_bf(v, aHi[m][k], aLo[m][k]);
      }
    }
    // cooperative fill: W tile, transposed to [n][k]
    #pragma unroll
    for (int idx = tid; idx < BN * BK; idx += 256) {
      const int n = idx >> 5, k = idx & 31;
      float v = W[(kc + k) * HID + (col0 + n)];
      split_bf(v, bHi[n][k], bLo[n][k]);
    }
    __syncthreads();

    union Frag { uint4 q[2]; v16bf v; };
    Frag ah, al;
    const int m = (wave << 4) + l15;
    ah.q[0] = *(const uint4*)&aHi[m][kh];
    ah.q[1] = *(const uint4*)&aHi[m][16 + kh];
    al.q[0] = *(const uint4*)&aLo[m][kh];
    al.q[1] = *(const uint4*)&aLo[m][16 + kh];

    #pragma unroll
    for (int nt = 0; nt < 4; ++nt) {
      Frag bh, bl;
      const int n = (nt << 4) + l15;
      bh.q[0] = *(const uint4*)&bHi[n][kh];
      bh.q[1] = *(const uint4*)&bHi[n][16 + kh];
      bl.q[0] = *(const uint4*)&bLo[n][kh];
      bl.q[1] = *(const uint4*)&bLo[n][16 + kh];
      // fp32-accurate product: hi*hi + hi*lo + lo*hi (lo*lo term ~2^-16, dropped)
      acc[nt] = __builtin_amdgcn_wmma_f32_16x16x32_bf16(false, ah.v, false, bh.v, (short)0, acc[nt], false, false);
      acc[nt] = __builtin_amdgcn_wmma_f32_16x16x32_bf16(false, ah.v, false, bl.v, (short)0, acc[nt], false, false);
      acc[nt] = __builtin_amdgcn_wmma_f32_16x16x32_bf16(false, al.v, false, bh.v, (short)0, acc[nt], false, false);
    }
  }

  // C/D layout: VGPR r -> row r (lanes 0-15) / row 8+r (lanes 16-31); N = lane%16
  const int rbase = row0 + (wave << 4) + ((lane & 16) ? 8 : 0);
  if (fullTile) {
    #pragma unroll
    for (int nt = 0; nt < 4; ++nt) {
      const int c = col0 + (nt << 4) + l15;
      #pragma unroll
      for (int r = 0; r < 8; ++r)
        C[(long long)(rbase + r) * HID + c] = acc[nt][r];
    }
  } else {
    for (int nt = 0; nt < 4; ++nt) {
      const int c = col0 + (nt << 4) + l15;
      for (int r = 0; r < 8; ++r) {
        const int row = rbase + r;
        if (row < nrows) C[(long long)row * HID + c] = acc[nt][r];
      }
    }
  }
}

// ---------- small GEMM for final 128 -> NCLS classes ----------
template <int CC>
__global__ void gemm_small(const float* __restrict__ h, const float* __restrict__ W,
                           float* __restrict__ out, int n) {
  const long long idx = (long long)blockIdx.x * blockDim.x + threadIdx.x;
  if (idx >= (long long)n * CC) return;
  const int node = (int)(idx / CC);
  const int c    = (int)(idx % CC);
  const float* hr = h + (long long)node * HID;
  float s = 0.0f;
  #pragma unroll 8
  for (int k = 0; k < HID; ++k) s += hr[k] * W[k * CC + c];
  out[idx] = s;
}

// ---------- graph preprocessing ----------
__global__ void fill_ones(float* p, int n) {
  int i = blockIdx.x * blockDim.x + threadIdx.x;
  if (i < n) p[i] = 1.0f;   // self-loop contribution to degree
}
__global__ void deg_accum(const int* __restrict__ dst, float* __restrict__ deg, int E) {
  int i = blockIdx.x * blockDim.x + threadIdx.x;
  if (i < E) atomicAdd(&deg[dst[i]], 1.0f);
}
__global__ void rsqrt_inplace(float* p, int n) {
  int i = blockIdx.x * blockDim.x + threadIdx.x;
  if (i < n) { float d = p[i]; p[i] = (d > 0.0f) ? rsqrtf(d) : 0.0f; }
}
__global__ void norm_kernel(const int* __restrict__ src, const int* __restrict__ dst,
                            const float* __restrict__ dinv, float* __restrict__ norm,
                            int E, int N) {
  int i = blockIdx.x * blockDim.x + threadIdx.x;
  if (i < E) { norm[i] = dinv[src[i]] * dinv[dst[i]]; }
  else if (i < E + N) { int v = i - E; norm[i] = dinv[v] * dinv[v]; }
}

// ---------- scatter: agg[dst] += tmp[src] * norm  (one wave32 per edge) ----------
template <int F>
__global__ void scatter_add(const float* __restrict__ tmp, const int* __restrict__ src,
                            const int* __restrict__ dst, const float* __restrict__ norm,
                            float* __restrict__ agg, int nE, int nN) {
  const int lane = threadIdx.x & 31;
  const long long item = (long long)blockIdx.x * (blockDim.x >> 5) + (threadIdx.x >> 5);
  if (item >= (long long)nE + nN) return;
  int s, d; float w;
  if (item < nE) { s = src[item]; d = dst[item]; w = norm[item]; }
  else           { int v = (int)(item - nE); s = v; d = v; w = norm[nE + v]; }
  const float* ts = tmp + (long long)s * F;
  float*       ad = agg + (long long)d * F;
  #pragma unroll
  for (int f = lane; f < F; f += 32)
    atomicAdd(&ad[f], ts[f] * w);
}

// ---------- elementwise ----------
__global__ void fzero(float* p, long long n) {
  long long i = (long long)blockIdx.x * blockDim.x + threadIdx.x;
  if (i < n) p[i] = 0.0f;
}
template <int F>
__global__ void bias_elu(float* __restrict__ h, const float* __restrict__ b, long long n) {
  long long i = (long long)blockIdx.x * blockDim.x + threadIdx.x;
  if (i >= n) return;
  float v = h[i] + b[(int)(i % F)];
  h[i] = (v > 0.0f) ? v : expm1f(v);
}
template <int F>
__global__ void bias_add(float* __restrict__ out, const float* __restrict__ b, long long n) {
  long long i = (long long)blockIdx.x * blockDim.x + threadIdx.x;
  if (i < n) out[i] += b[(int)(i % F)];
}

static inline int cdiv(long long a, long long b) { return (int)((a + b - 1) / b); }

extern "C" void kernel_launch(void* const* d_in, const int* in_sizes, int n_in,
                              void* d_out, int out_size, void* d_ws, size_t ws_size,
                              hipStream_t stream) {
  (void)n_in; (void)out_size; (void)ws_size;
  const float* x    = (const float*)d_in[0];
  const int*   eidx = (const int*)d_in[1];
  const float* Wst  = (const float*)d_in[2];
  const float* bst  = (const float*)d_in[3];
  const float* Wout = (const float*)d_in[4];
  const float* bout = (const float*)d_in[5];

  const int N = in_sizes[0] / HID;  // 50000 nodes
  const int E = in_sizes[1] / 2;    // 600000 edges
  constexpr int NCLS = 10;          // fixed by reference (N_CLASSES)

  const int* src = eidx;
  const int* dst = eidx + E;

  // workspace layout (floats): ~54 MB total
  float* norm = (float*)d_ws;                         // E + N
  float* deg  = norm + ((size_t)E + N);               // N (becomes dinv in-place)
  float* hbuf = deg + N;                              // N * HID (aggregation / h)
  float* tmp  = hbuf + (size_t)N * HID;               // N * HID (GEMM output)

  // --- graph normalization (deg on dst + self-loops; norm = dinv[s]*dinv[d]) ---
  fill_ones<<<cdiv(N, 256), 256, 0, stream>>>(deg, N);
  deg_accum<<<cdiv(E, 256), 256, 0, stream>>>(dst, deg, E);
  rsqrt_inplace<<<cdiv(N, 256), 256, 0, stream>>>(deg, N);
  norm_kernel<<<cdiv((long long)E + N, 256), 256, 0, stream>>>(src, dst, deg, norm, E, N);

  const long long nh    = (long long)N * HID;
  const int scat_blocks = cdiv((long long)E + N, 8);  // 8 wave32 edges per 256-thr block
  const dim3 ggrid(cdiv(N, BM), HID / BN);

  // --- 8 GCN layers with ELU ---
  const float* hin = x;
  for (int l = 0; l < 8; ++l) {
    gemm_bf16x3<<<ggrid, 256, 0, stream>>>(hin, Wst + (size_t)l * HID * HID, tmp, N);
    fzero<<<cdiv(nh, 256), 256, 0, stream>>>(hbuf, nh);
    scatter_add<HID><<<scat_blocks, 256, 0, stream>>>(tmp, src, dst, norm, hbuf, E, N);
    bias_elu<HID><<<cdiv(nh, 256), 256, 0, stream>>>(hbuf, bst + (size_t)l * HID, nh);
    hin = hbuf;
  }

  // --- output layer: 128 -> NCLS, aggregate, bias (no activation) ---
  const long long no = (long long)N * NCLS;
  gemm_small<NCLS><<<cdiv(no, 256), 256, 0, stream>>>(hbuf, Wout, tmp, N);
  fzero<<<cdiv(no, 256), 256, 0, stream>>>((float*)d_out, no);
  scatter_add<NCLS><<<scat_blocks, 256, 0, stream>>>(tmp, src, dst, norm, (float*)d_out, E, N);
  bias_add<NCLS><<<cdiv(no, 256), 256, 0, stream>>>((float*)d_out, bout, no);
}
